// MPC_ExternalAttention_47330539602295
// MI455X (gfx1250) — matrix operand order repaired and checked by reading, not compiled, not run on hardware
//
#include <hip/hip_runtime.h>

// External attention: out = normalize((x @ Wk + 5)^2) @ Wv
//   x  : [16, 8192, 256] f32  -> M = 131072 rows, D = 256
//   Wk : [256, 128] f32  (D x S)
//   Wv : [128, 256] f32  (S x D)
// Strategy: split-precision bf16 WMMA (hi/lo decomposition, 3 products per
// tile) on v_wmma_f32_16x16x32_bf16. One wave (32 threads) per workgroup,
// 16 output rows per wave, no inter-wave sync needed.

#define DD 256
#define SS 128
#define CADD 5.0f

typedef __attribute__((ext_vector_type(16))) __bf16 v16bf;
typedef __attribute__((ext_vector_type(8)))  __bf16 v8bf;
typedef __attribute__((ext_vector_type(8)))  float  v8f;
typedef __attribute__((ext_vector_type(4)))  float  f32x4;

static __device__ __forceinline__ v16bf concat8(v8bf a, v8bf b) {
  return __builtin_shufflevector(a, b, 0,1,2,3,4,5,6,7,8,9,10,11,12,13,14,15);
}

// ---------------------------------------------------------------------------
// Prep: build transposed bf16 hi/lo copies of the weights in workspace.
//   WkT_hi/lo : [S=128][D=256]  (column-of-Wk contiguous = K-contiguous B)
//   WvT_hi/lo : [D=256][S=128]  (column-of-Wv contiguous = K-contiguous B)
// ---------------------------------------------------------------------------
__global__ __launch_bounds__(256)
void prep_weights(const float* __restrict__ Wk, const float* __restrict__ Wv,
                  __bf16* __restrict__ WkT_hi, __bf16* __restrict__ WkT_lo,
                  __bf16* __restrict__ WvT_hi, __bf16* __restrict__ WvT_lo) {
  int idx = blockIdx.x * 256 + threadIdx.x;      // 0..32767
  // WkT[s][d] = Wk[d][s]
  int s = idx >> 8;        // /256
  int d = idx & 255;
  float wk = Wk[d * SS + s];
  __bf16 kh = (__bf16)wk;
  WkT_hi[idx] = kh;
  WkT_lo[idx] = (__bf16)(wk - (float)kh);
  // WvT[d2][s2] = Wv[s2][d2]
  int d2 = idx >> 7;       // /128
  int s2 = idx & 127;
  float wv = Wv[s2 * DD + d2];
  __bf16 vh = (__bf16)wv;
  WvT_hi[idx] = vh;
  WvT_lo[idx] = (__bf16)(wv - (float)vh);
}

// ---------------------------------------------------------------------------
// Main kernel: one wave per block, 16 rows per block.
// ---------------------------------------------------------------------------
__global__ __launch_bounds__(32)
void ext_attn_kernel(const float* __restrict__ x,
                     const __bf16* __restrict__ WkT_hi,
                     const __bf16* __restrict__ WkT_lo,
                     const __bf16* __restrict__ WvT_hi,
                     const __bf16* __restrict__ WvT_lo,
                     float* __restrict__ out) {
  // P staged in LDS to convert C-layout -> A-layout between the GEMMs.
  // Row stride 136 (=128+8) bf16 keeps b128 A-loads on distinct banks.
  __shared__ __align__(16) __bf16 Phi[16 * 136];
  __shared__ __align__(16) __bf16 Plo[16 * 136];

  const int lane = threadIdx.x;        // 0..31
  const int h    = lane >> 4;          // lane half (K-subset select)
  const int n16  = lane & 15;          // row (A) / column (B,C) index
  const long row0 = (long)blockIdx.x * 16;

  // ---------------- GEMM1: attn = x @ Wk  (+5 via accumulator init) -------
  v8f c[8];
  #pragma unroll
  for (int t = 0; t < 8; ++t)
    #pragma unroll
    for (int r = 0; r < 8; ++r) c[t][r] = CADD;

  const float* xrow = x + (row0 + n16) * DD;
  for (int kk = 0; kk < 8; ++kk) {
    const int k = kk * 32;
    // A tile 16x32: lane holds its row's K in [k+8h, k+8h+8) U [k+16+8h, +8)
    f32x4 f0 = *(const f32x4*)(xrow + k + 8 * h);
    f32x4 f1 = *(const f32x4*)(xrow + k + 8 * h + 4);
    f32x4 f2 = *(const f32x4*)(xrow + k + 16 + 8 * h);
    f32x4 f3 = *(const f32x4*)(xrow + k + 16 + 8 * h + 4);
    float xv[16] = { f0[0], f0[1], f0[2], f0[3],  f1[0], f1[1], f1[2], f1[3],
                     f2[0], f2[1], f2[2], f2[3],  f3[0], f3[1], f3[2], f3[3] };
    v16bf ahi, alo;
    #pragma unroll
    for (int i = 0; i < 16; ++i) {
      float f  = xv[i];
      __bf16 hb = (__bf16)f;
      ahi[i] = hb;
      alo[i] = (__bf16)(f - (float)hb);
    }
    #pragma unroll
    for (int t = 0; t < 8; ++t) {
      // B tile 32x16: lane holds column (t*16+n16), K in [k+8h, k+8h+8) U [k+16+8h, +8)
      const __bf16* bh = WkT_hi + (t * 16 + n16) * DD + k + 8 * h;
      const __bf16* bl = WkT_lo + (t * 16 + n16) * DD + k + 8 * h;
      v16bf bhi = concat8(*(const v8bf*)bh, *(const v8bf*)(bh + 16));
      v16bf blo = concat8(*(const v8bf*)bl, *(const v8bf*)(bl + 16));
      c[t] = __builtin_amdgcn_wmma_f32_16x16x32_bf16(false, ahi, false, bhi,
                                                     (short)0, c[t], false, false);
      c[t] = __builtin_amdgcn_wmma_f32_16x16x32_bf16(false, ahi, false, blo,
                                                     (short)0, c[t], false, false);
      c[t] = __builtin_amdgcn_wmma_f32_16x16x32_bf16(false, alo, false, bhi,
                                                     (short)0, c[t], false, false);
    }
  }

  // ---------------- Quad2 normalize: (a+5)^2 / rowsum ----------------------
  float rs[8];
  #pragma unroll
  for (int r = 0; r < 8; ++r) rs[r] = 0.0f;
  #pragma unroll
  for (int t = 0; t < 8; ++t)
    #pragma unroll
    for (int r = 0; r < 8; ++r) {
      float v = c[t][r];   // = dot + 5
      v *= v;
      c[t][r] = v;
      rs[r] += v;
    }
  // C layout: reg r, lane L -> row (r + 8h), col (L&15). Reduce over the 16
  // lanes of each half (xor 1,2,4,8 stays inside the half).
  #pragma unroll
  for (int r = 0; r < 8; ++r) {
    float s = rs[r];
    s += __shfl_xor(s, 1, 32);
    s += __shfl_xor(s, 2, 32);
    s += __shfl_xor(s, 4, 32);
    s += __shfl_xor(s, 8, 32);
    rs[r] = 1.0f / s;
  }
  // Store normalized P as bf16 hi/lo in row-major LDS (C-layout -> A-layout).
  #pragma unroll
  for (int t = 0; t < 8; ++t)
    #pragma unroll
    for (int r = 0; r < 8; ++r) {
      float p = c[t][r] * rs[r];
      __bf16 ph = (__bf16)p;
      __bf16 pl = (__bf16)(p - (float)ph);
      int a = (r + 8 * h) * 136 + t * 16 + n16;
      Phi[a] = ph;
      Plo[a] = pl;
    }
  // Same wave reads back its own writes; compiler inserts s_wait_dscnt.

  // ---------------- GEMM2: out = P @ Wv -----------------------------------
  #pragma unroll 1
  for (int dc = 0; dc < 2; ++dc) {
    v8f o[8];
    #pragma unroll
    for (int t = 0; t < 8; ++t)
      #pragma unroll
      for (int r = 0; r < 8; ++r) o[t][r] = 0.0f;

    for (int ks = 0; ks < 4; ++ks) {
      const int k = ks * 32;
      const __bf16* pa = &Phi[n16 * 136 + k + 8 * h];
      const __bf16* pb = &Plo[n16 * 136 + k + 8 * h];
      v16bf ahi = concat8(*(const v8bf*)pa, *(const v8bf*)(pa + 16));
      v16bf alo = concat8(*(const v8bf*)pb, *(const v8bf*)(pb + 16));
      #pragma unroll
      for (int t = 0; t < 8; ++t) {
        const __bf16* bh = WvT_hi + (dc * 128 + t * 16 + n16) * SS + k + 8 * h;
        const __bf16* bl = WvT_lo + (dc * 128 + t * 16 + n16) * SS + k + 8 * h;
        v16bf bhi = concat8(*(const v8bf*)bh, *(const v8bf*)(bh + 16));
        v16bf blo = concat8(*(const v8bf*)bl, *(const v8bf*)(bl + 16));
        o[t] = __builtin_amdgcn_wmma_f32_16x16x32_bf16(false, ahi, false, bhi,
                                                       (short)0, o[t], false, false);
        o[t] = __builtin_amdgcn_wmma_f32_16x16x32_bf16(false, ahi, false, blo,
                                                       (short)0, o[t], false, false);
        o[t] = __builtin_amdgcn_wmma_f32_16x16x32_bf16(false, alo, false, bhi,
                                                       (short)0, o[t], false, false);
      }
    }
    #pragma unroll
    for (int t = 0; t < 8; ++t)
      #pragma unroll
      for (int r = 0; r < 8; ++r)
        out[(row0 + r + 8 * h) * DD + dc * 128 + t * 16 + n16] = o[t][r];
  }
}

// ---------------------------------------------------------------------------
extern "C" void kernel_launch(void* const* d_in, const int* in_sizes, int n_in,
                              void* d_out, int out_size, void* d_ws, size_t ws_size,
                              hipStream_t stream) {
  const float* x  = (const float*)d_in[0];
  const float* Wk = (const float*)d_in[1];
  const float* Wv = (const float*)d_in[2];
  float* out = (float*)d_out;

  // Workspace layout (bf16 weight copies, 256 KB total):
  char* ws = (char*)d_ws;
  __bf16* WkT_hi = (__bf16*)(ws);
  __bf16* WkT_lo = (__bf16*)(ws + 65536);
  __bf16* WvT_hi = (__bf16*)(ws + 131072);
  __bf16* WvT_lo = (__bf16*)(ws + 196608);

  prep_weights<<<128, 256, 0, stream>>>(Wk, Wv, WkT_hi, WkT_lo, WvT_hi, WvT_lo);

  const int M = in_sizes[0] / DD;               // 131072 rows
  ext_attn_kernel<<<M / 16, 32, 0, stream>>>(x, WkT_hi, WkT_lo, WvT_hi, WvT_lo, out);
}
